// MultiHead_50577534878152
// MI455X (gfx1250) — matrix operand order repaired and checked by reading, not compiled, not run on hardware
//
#include <hip/hip_runtime.h>

// Fused pre-norm multi-head attention block for MI455X (gfx1250, wave32).
// One workgroup (128 threads = 4 wave32) per batch element; everything after
// the initial Q/K/V stream stays in LDS. GEMMs use v_wmma_f32_16x16x32_f16.
// Q/K/V are streamed with GLOBAL_LOAD_ASYNC_TO_LDS_B128 (ASYNCcnt), software-
// pipelined so the V stream overlaps the K projection WMMA work.

namespace {

constexpr int S_  = 50;   // sequence length
constexpr int D_  = 48;   // model dim
constexpr int H_  = 3;    // heads
constexpr int HD_ = 16;   // head dim
constexpr int SP_ = 64;   // padded sequence (4 x 16 M-tiles)
constexpr int DP_ = 64;   // padded model dim (2 x 32 K-steps)
constexpr float NEG_ = -1e9f;

typedef __attribute__((ext_vector_type(16))) _Float16 v16h;
typedef __attribute__((ext_vector_type(8)))  float    v8f;

union Frag16 { v16h v; unsigned int u[8]; };

__device__ inline v8f wmma32(v16h a, v16h b, v8f c) {
  // D = A(16x32 f16) * B(32x16 f16) + C(16x16 f32)
  return __builtin_amdgcn_wmma_f32_16x16x32_f16(
      /*neg_a=*/false, a, /*neg_b=*/false, b,
      /*c_mod=*/(short)0, c, /*reuse_a=*/false, /*reuse_b=*/false);
}

// Low 32 bits of a generic pointer into LDS == byte offset within the
// workgroup's LDS allocation (aperture base lives in the high bits).
__device__ inline unsigned lds_off(const void* p) {
  return (unsigned)(reinterpret_cast<uintptr_t>(p));
}

// GLOBAL_LOAD_ASYNC_TO_LDS_B128, GVS mode: mem = SGPR base + 32-bit VGPR off.
// VDST slot carries the LDS byte address. Tracked by ASYNCcnt.
__device__ inline void async_copy_b128(unsigned lds_addr, const void* gbase,
                                       unsigned goff) {
  asm volatile("global_load_async_to_lds_b128 %0, %1, %2"
               :: "v"(lds_addr), "v"(goff), "s"(gbase)
               : "memory");
}

__device__ inline void wait_async0() {
  asm volatile("s_wait_asynccnt 0x0" ::: "memory");
}

// A fragment from row-major f16 LDS (M=16 x K=32 tile at A, row stride lda halves).
// Lane l holds row m=l&15; element i maps to k = 16*hi + i -> contiguous.
__device__ inline v16h load_a(const _Float16* A, int lda, int kbase, int lane) {
  const int m = lane & 15, hi = lane >> 4;
  const unsigned int* row = (const unsigned int*)(A + m * lda + kbase);
  Frag16 f;
#pragma unroll
  for (int g = 0; g < 2; ++g)
#pragma unroll
    for (int j = 0; j < 4; ++j)
      f.u[4 * g + j] = row[8 * hi + 4 * g + j];
  return f.v;
}

// B fragment where LDS holds B transposed ("n-major": BT[n][k], row stride ldb).
// Lane l holds col n=l&15; element i maps to k = kbase + 16*hi + i -> contiguous.
__device__ inline v16h load_bt(const _Float16* BT, int ldb, int kbase, int lane) {
  const int n = lane & 15, hi = lane >> 4;
  const unsigned int* row = (const unsigned int*)(BT + n * ldb + kbase + hi * 16);
  Frag16 f;
#pragma unroll
  for (int j = 0; j < 8; ++j) f.u[j] = row[j];
  return f.v;
}

// K=16 variants (head-dim GEMMs): zero-fill the k>=16 half of the K=32 fragment.
__device__ inline v16h load_a_k16(const _Float16* A, int lda, int lane) {
  const int m = lane & 15, hi = lane >> 4;
  const unsigned int* row = (const unsigned int*)(A + m * lda + 8 * hi);
  Frag16 f;
#pragma unroll
  for (int j = 0; j < 4; ++j) f.u[j] = row[j];
#pragma unroll
  for (int j = 4; j < 8; ++j) f.u[j] = 0u;
  return f.v;
}

__device__ inline v16h load_bt_k16(const _Float16* BT, int ldb, int lane) {
  const int n = lane & 15, hi = lane >> 4;
  const unsigned int* row = (const unsigned int*)(BT + n * ldb + 8 * hi);
  Frag16 f;
#pragma unroll
  for (int j = 0; j < 8; ++j) f.u[j] = (j < 4) ? row[j] : 0u;
  return f.v;
}

__global__ void __launch_bounds__(128, 1)
mha_fused_kernel(const float* __restrict__ Qg, const float* __restrict__ Kg,
                 const float* __restrict__ Vg, const unsigned char* __restrict__ Mg,
                 const float* __restrict__ Wq, const float* __restrict__ bq,
                 const float* __restrict__ Wk, const float* __restrict__ bk,
                 const float* __restrict__ Wv, const float* __restrict__ bv,
                 const float* __restrict__ Wo, const float* __restrict__ bo,
                 const float* __restrict__ gamma, const float* __restrict__ beta,
                 float* __restrict__ Og)
{
  __shared__ __align__(16) float    s_resid[S_ * D_];  // original Q (residual), 9.6 KB
  __shared__ __align__(16) float    s_f32[S_ * DP_];   // K/V staging then scores, 12.8 KB
  __shared__ _Float16 s_h16[SP_ * DP_];        // LN output, then attn weights, 8 KB
  __shared__ _Float16 s_w[D_ * DP_];           // current weight f16, zero K-pad, 6 KB
  __shared__ _Float16 s_q[H_ * SP_ * HD_];     // q per head, row-major, 6 KB
  __shared__ _Float16 s_k[H_ * SP_ * HD_];     // k per head, row-major (=scores BT), 6 KB
  __shared__ _Float16 s_vT[H_ * HD_ * SP_];    // v per head, transposed (=ctx BT), 6 KB
  __shared__ _Float16 s_ctx[SP_ * DP_];        // context, zero K-pad, 8 KB

  const int tid  = threadIdx.x;
  const int wave = tid >> 5;
  const int lane = tid & 31;
  const int bidx = blockIdx.x;
  const size_t xoff = (size_t)bidx * (S_ * D_);

  // S*D floats = 9600 B = 600 x 16 B chunks per tensor.
  // Q -> s_resid, densely packed (row stride 48 floats = 192 B = 12 chunks).
  auto issue_async_dense = [&](const float* X) {
    const void* gb = (const void*)(X + xoff);
    const unsigned lbase = lds_off(s_resid);
    for (int i = tid; i < 600; i += 128)
      async_copy_b128(lbase + (unsigned)i * 16u, gb, (unsigned)i * 16u);
  };
  // K/V -> s_f32 with row padding to DP_ (row: 192 B global -> 256 B LDS).
  auto issue_async_padded = [&](const float* X) {
    const void* gb = (const void*)(X + xoff);
    const unsigned lbase = lds_off(s_f32);
    for (int i = tid; i < 600; i += 128) {
      const unsigned r = (unsigned)i / 12u, c = (unsigned)i % 12u;
      async_copy_b128(lbase + r * 256u + c * 16u, gb, (unsigned)i * 16u);
    }
  };

  auto load_weight = [&](const float* W) {
    for (int i = tid; i < D_ * DP_; i += 128) {
      const int r = i >> 6, c = i & 63;
      s_w[i] = (c < D_) ? (_Float16)W[r * D_ + c] : (_Float16)0.f;
    }
  };

  auto layer_norm = [&](const float* src, int stride) {
    if (tid < S_) {
      const float* row = src + tid * stride;
      float mu = 0.f;
      for (int c = 0; c < D_; ++c) mu += row[c];
      mu *= (1.f / D_);
      float var = 0.f;
      for (int c = 0; c < D_; ++c) { const float d = row[c] - mu; var += d * d; }
      var *= (1.f / D_);
      const float rs = rsqrtf(var + 1e-5f);
      for (int c = 0; c < D_; ++c)
        s_h16[tid * DP_ + c] = (_Float16)((row[c] - mu) * rs * gamma[c] + beta[c]);
      for (int c = D_; c < DP_; ++c) s_h16[tid * DP_ + c] = (_Float16)0.f;
    } else if (tid < SP_) {
      for (int c = 0; c < DP_; ++c) s_h16[tid * DP_ + c] = (_Float16)0.f;
    }
  };

  // out = Xn @ W^T + bias; 12 tiles (4 M x 3 N) over 4 waves; dest: 0=q,1=k,2=vT
  auto project = [&](const float* bias, int dest) {
    for (int t = wave; t < 12; t += 4) {
      const int mt = t & 3, nt = t >> 2;          // nt == head
      const int n = lane & 15, hi = lane >> 4;
      const float bval = bias[nt * 16 + n];
      v8f acc;
#pragma unroll
      for (int r = 0; r < 8; ++r) acc[r] = bval;  // bias depends on n only
#pragma unroll
      for (int ks = 0; ks < 2; ++ks) {
        v16h a = load_a(s_h16 + mt * 16 * DP_, DP_, ks * 32, lane);
        v16h b = load_bt(s_w + nt * 16 * DP_, DP_, ks * 32, lane);  // W row-major == B^T
        acc = wmma32(a, b, acc);
      }
#pragma unroll
      for (int r = 0; r < 8; ++r) {
        const int m = mt * 16 + r + 8 * hi;
        const _Float16 hv = (_Float16)acc[r];
        if (dest == 0)      s_q[(nt * SP_ + m) * HD_ + n] = hv;
        else if (dest == 1) s_k[(nt * SP_ + m) * HD_ + n] = hv;
        else                s_vT[(nt * HD_ + n) * SP_ + m] = hv;
      }
    }
  };

  // ---- phase A: kick off Q and K streams, stage Wq, zero ctx padding -------
  issue_async_dense(Qg);       // Q -> s_resid
  issue_async_padded(Kg);      // K -> s_f32 (overlaps ctx zeroing + Wq staging)
  for (int i = tid; i < SP_ * DP_; i += 128) s_ctx[i] = (_Float16)0.f;
  load_weight(Wq);
  wait_async0();
  __syncthreads();

  // ---- phase B: LN(Q); also cache this thread's mask row as a bitmap ------
  unsigned long long mbits = 0ull;             // reused for all 3 heads
  if (tid < S_) {
    const unsigned char* mrow = Mg + (size_t)bidx * (S_ * S_) + (size_t)tid * S_;
    for (int k = 0; k < S_; ++k)
      mbits |= (unsigned long long)(mrow[k] != 0) << k;
  }
  layer_norm(s_resid, D_);
  __syncthreads();

  project(bq, 0);
  __syncthreads();

  // ---- K: LN + projection; V stream issued under the k-projection ----------
  load_weight(Wk);
  layer_norm(s_f32, DP_);
  __syncthreads();

  issue_async_padded(Vg);      // V -> s_f32 (s_f32 free: LN(K) consumed it)
  project(bk, 1);
  wait_async0();
  __syncthreads();

  load_weight(Wv);
  layer_norm(s_f32, DP_);
  __syncthreads();

  project(bv, 2);
  __syncthreads();

  load_weight(Wo);             // s_w free from here until the output projection
  __syncthreads();

  // ---- attention per head --------------------------------------------------
  for (int h = 0; h < H_; ++h) {
    // scores = q @ k^T * (1/sqrt(HD)); 16 tiles (4x4) over 4 waves, K=16
    for (int t = wave; t < 16; t += 4) {
      const int mt = t & 3, nt = t >> 2;
      const int n = lane & 15, hi = lane >> 4;
      v8f acc;
#pragma unroll
      for (int r = 0; r < 8; ++r) acc[r] = 0.f;
      v16h a  = load_a_k16(s_q + (h * SP_ + mt * 16) * HD_, HD_, lane);
      v16h bb = load_bt_k16(s_k + (h * SP_ + nt * 16) * HD_, HD_, lane);
      acc = wmma32(a, bb, acc);
#pragma unroll
      for (int r = 0; r < 8; ++r) {
        const int m = mt * 16 + r + 8 * hi;
        if (m < S_) s_f32[m * DP_ + nt * 16 + n] = acc[r] * 0.25f;  // 1/sqrt(16)
      }
    }
    __syncthreads();

    // masked softmax, one query row per thread; result as f16 A-matrix
    if (tid < S_) {
      float* srow = s_f32 + tid * DP_;
      float mx = -3.402823466e38f;
      for (int k = 0; k < S_; ++k) {
        const float v = ((mbits >> k) & 1ull) ? NEG_ : srow[k];
        mx = fmaxf(mx, v);
      }
      float sum = 0.f;
      for (int k = 0; k < S_; ++k) {
        const float v = ((mbits >> k) & 1ull) ? NEG_ : srow[k];
        const float e = __expf(v - mx);
        sum += e;
        srow[k] = e;
      }
      const float inv = 1.f / sum;
      for (int k = 0; k < S_; ++k) s_h16[tid * DP_ + k] = (_Float16)(srow[k] * inv);
      for (int k = S_; k < DP_; ++k) s_h16[tid * DP_ + k] = (_Float16)0.f;
    } else if (tid < SP_) {
      for (int k = 0; k < DP_; ++k) s_h16[tid * DP_ + k] = (_Float16)0.f;
    }
    __syncthreads();

    // ctx = attn @ v; 4 M-tiles, one per wave; K = 64 (padded keys are zero attn)
    {
      const int mt = wave;
      const int n = lane & 15, hi = lane >> 4;
      v8f acc;
#pragma unroll
      for (int r = 0; r < 8; ++r) acc[r] = 0.f;
#pragma unroll
      for (int ks = 0; ks < 2; ++ks) {
        v16h a  = load_a(s_h16 + mt * 16 * DP_, DP_, ks * 32, lane);
        v16h bb = load_bt(s_vT + h * HD_ * SP_, SP_, ks * 32, lane);
        acc = wmma32(a, bb, acc);
      }
#pragma unroll
      for (int r = 0; r < 8; ++r) {
        const int m = mt * 16 + r + 8 * hi;
        s_ctx[m * DP_ + h * 16 + n] = (_Float16)acc[r];
      }
    }
    __syncthreads();
  }

  // ---- output projection + residual ---------------------------------------
  for (int t = wave; t < 12; t += 4) {
    const int mt = t & 3, nt = t >> 2;
    const int n = lane & 15, hi = lane >> 4;
    const int col = nt * 16 + n;
    v8f acc;
#pragma unroll
    for (int r = 0; r < 8; ++r) acc[r] = bo[col];
#pragma unroll
    for (int ks = 0; ks < 2; ++ks) {
      v16h a  = load_a(s_ctx + mt * 16 * DP_, DP_, ks * 32, lane);
      v16h bb = load_bt(s_w + nt * 16 * DP_, DP_, ks * 32, lane);
      acc = wmma32(a, bb, acc);
    }
#pragma unroll
    for (int r = 0; r < 8; ++r) {
      const int m = mt * 16 + r + 8 * hi;
      if (m < S_) Og[xoff + m * D_ + col] = acc[r] + s_resid[m * D_ + col];
    }
  }
}

} // namespace

extern "C" void kernel_launch(void* const* d_in, const int* in_sizes, int n_in,
                              void* d_out, int out_size, void* d_ws, size_t ws_size,
                              hipStream_t stream) {
  (void)n_in; (void)out_size; (void)d_ws; (void)ws_size;
  const float* Qg = (const float*)d_in[0];
  const float* Kg = (const float*)d_in[1];
  const float* Vg = (const float*)d_in[2];
  const unsigned char* Mg = (const unsigned char*)d_in[3];  // bool mask (b,1,S,S)
  const float* Wq = (const float*)d_in[4];
  const float* bq = (const float*)d_in[5];
  const float* Wk = (const float*)d_in[6];
  const float* bk = (const float*)d_in[7];
  const float* Wv = (const float*)d_in[8];
  const float* bv = (const float*)d_in[9];
  const float* Wo = (const float*)d_in[10];
  const float* bo = (const float*)d_in[11];
  const float* gamma = (const float*)d_in[12];
  const float* beta  = (const float*)d_in[13];
  float* Og = (float*)d_out;

  const int b = in_sizes[0] / (S_ * D_);  // 8192
  mha_fused_kernel<<<dim3(b), dim3(128), 0, stream>>>(
      Qg, Kg, Vg, Mg, Wq, bq, Wk, bk, Wv, bv, Wo, bo, gamma, beta, Og);
}
